// Siglip2VisionEmbeddings_4303557231407
// MI455X (gfx1250) — matrix operand order repaired and checked by reading, not compiled, not run on hardware
//
#include <hip/hip_runtime.h>

typedef unsigned short u16;
typedef unsigned int u32;

typedef __attribute__((ext_vector_type(16))) __bf16 v16bf;
typedef __attribute__((ext_vector_type(8)))  float  v8f;

#define BQ     64
#define LQ     1024
#define KDIM   768
#define DDIM   1152
#define GRID16 16
#define MROWS  (BQ * LQ)

#define BM 128
#define BN 128
#define BK 32
#define LDA 40   // padded LDS row stride (shorts): 80B = 20 banks -> conflict-free b128
#define LDB 40
#define KT (KDIM / BK)  // 24

// fp32 -> bf16 (round-half-up), two at a time:
// 2x v_add + 1x v_perm_b32 — no half-register shuffling.
__device__ __forceinline__ u32 pack_bf16(float a, float b) {
  u32 ua = __float_as_uint(a) + 0x8000u;
  u32 ub = __float_as_uint(b) + 0x8000u;
  // result bytes: [0,1] = ua bytes [2,3] (low half = bf16(a)), [2,3] = ub bytes [2,3]
  return __builtin_amdgcn_perm(ub, ua, 0x07060302u);
}

// generic LDS pointer -> 32-bit LDS byte offset (flat shared addr low bits)
__device__ __forceinline__ u32 lds_off(const void* p) {
  return (u32)(unsigned long long)(uintptr_t)p;
}

// 16B async copy global -> LDS, per lane (ASYNCcnt-tracked, no VGPR data)
__device__ __forceinline__ void async_copy_b128(u32 ldsByteOff, const void* gaddr) {
  asm volatile("global_load_async_to_lds_b128 %0, %1, off"
               :: "v"(ldsByteOff), "v"(gaddr) : "memory");
}

__device__ __forceinline__ void wait_asynccnt0() {
  asm volatile("s_wait_asynccnt 0x0" ::: "memory");
}

// --- one-time W fp32 -> bf16 (1152x768), 8 elements / thread ---------------
__global__ void prep_w_kernel(const float* __restrict__ W, u32* __restrict__ Wbf) {
  const int i = blockIdx.x * 256 + threadIdx.x;   // 110592 threads exactly
  const float4* src = reinterpret_cast<const float4*>(W) + (size_t)i * 2;
  float4 a = src[0], b = src[1];
  uint4 o;
  o.x = pack_bf16(a.x, a.y);
  o.y = pack_bf16(a.z, a.w);
  o.z = pack_bf16(b.x, b.y);
  o.w = pack_bf16(b.z, b.w);
  reinterpret_cast<uint4*>(Wbf)[i] = o;
}

// --- per-row bilinear params: 4 indices + 4 weights per (b,l) --------------
__global__ void prep_pos_kernel(const int* __restrict__ ss,
                                int4* __restrict__ rp_idx,
                                float4* __restrict__ rp_w) {
  const int r = blockIdx.x * 256 + threadIdx.x;
  if (r >= MROWS) return;
  const int b = r >> 10;
  int l = r & (LQ - 1);
  const int h = ss[2 * b];
  const int w = ss[2 * b + 1];
  if (l >= h * w) l = 0;                 // pad rows replicate interpolated (0,0)
  const int y = l / w;
  const int x = l - y * w;
  // half-pixel centers (align_corners=False)
  const float fy = ((float)y + 0.5f) * ((float)GRID16 / (float)h) - 0.5f;
  const float fx = ((float)x + 0.5f) * ((float)GRID16 / (float)w) - 0.5f;
  const float fy0 = floorf(fy), fx0 = floorf(fx);
  const float ty = fy - fy0, tx = fx - fx0;
  const int y0 = (int)fy0, x0 = (int)fx0;
  const int iy0 = min(max(y0, 0), GRID16 - 1);
  const int iy1 = min(max(y0 + 1, 0), GRID16 - 1);
  const int ix0 = min(max(x0, 0), GRID16 - 1);
  const int ix1 = min(max(x0 + 1, 0), GRID16 - 1);
  int4 idx;
  idx.x = iy0 * GRID16 + ix0;
  idx.y = iy0 * GRID16 + ix1;
  idx.z = iy1 * GRID16 + ix0;
  idx.w = iy1 * GRID16 + ix1;
  float4 wt;
  wt.x = (1.f - ty) * (1.f - tx);
  wt.y = (1.f - ty) * tx;
  wt.z = ty * (1.f - tx);
  wt.w = ty * tx;
  rp_idx[r] = idx;
  rp_w[r] = wt;
}

// --- fused GEMM + bias + bilinear pos add ----------------------------------
__global__ __launch_bounds__(256) void embed_gemm_kernel(
    const float* __restrict__ pix,    // [M, K] fp32
    const u16*  __restrict__ Wbf,     // [D, K] bf16
    const float* __restrict__ bias,   // [D]
    const float* __restrict__ pos,    // [256, D] fp32 (L2-resident)
    const int4* __restrict__ rp_idx,  // [M]
    const float4* __restrict__ rp_w,  // [M]
    float* __restrict__ out)          // [M, D]
{
  __shared__ u16 As[2][BM * LDA];
  __shared__ u16 Bs[2][BN * LDB];

  const int tid   = threadIdx.x;
  const int lane  = tid & 31;
  const int half  = lane >> 4;
  const int ln    = lane & 15;
  const int wid   = tid >> 5;
  const int waveM = wid & 3;   // 4 M-strips of 32
  const int waveN = wid >> 2;  // 2 N-strips of 64

  // N-tile varies fastest across the grid so the 9 blocks sharing one
  // 128-row A band dispatch adjacently -> A is HBM-read once, L2-served 9x.
  const int gn0 = blockIdx.x * BN;
  const int gm0 = blockIdx.y * BM;

  // cooperative copy assignment: 2 threads per row, 16 K-elems each
  const int crow = tid >> 1;
  const int cseg = (tid & 1) << 4;

  const float* aSrcBase = pix + (size_t)(gm0 + crow) * KDIM + cseg;
  const u16*   bSrcBase = Wbf + (size_t)(gn0 + crow) * KDIM + cseg;

  v8f acc[2][4] = {};

  // --- stage k-tile 0 into buffer 0 ---
  {
    // B tile: pure bf16 copy -> async global->LDS (no VGPR round trip)
    const u32 boff = lds_off(&Bs[0][crow * LDB + cseg]);
    async_copy_b128(boff,      bSrcBase);
    async_copy_b128(boff + 16, bSrcBase + 8);
    // A tile: fp32 load + pack -> LDS
    const float4* s = reinterpret_cast<const float4*>(aSrcBase);
    float4 f0 = s[0], f1 = s[1], f2 = s[2], f3 = s[3];
    uint4 p0, p1;
    p0.x = pack_bf16(f0.x, f0.y); p0.y = pack_bf16(f0.z, f0.w);
    p0.z = pack_bf16(f1.x, f1.y); p0.w = pack_bf16(f1.z, f1.w);
    p1.x = pack_bf16(f2.x, f2.y); p1.y = pack_bf16(f2.z, f2.w);
    p1.z = pack_bf16(f3.x, f3.y); p1.w = pack_bf16(f3.z, f3.w);
    u16* d = &As[0][crow * LDA + cseg];
    *reinterpret_cast<uint4*>(d)     = p0;
    *reinterpret_cast<uint4*>(d + 8) = p1;
    wait_asynccnt0();
  }
  __syncthreads();

  for (int kt = 0; kt < KT; ++kt) {
    const int cur = kt & 1;
    const int nxt = cur ^ 1;
    const bool hasNext = (kt + 1) < KT;

    float4 f0, f1, f2, f3;
    if (hasNext) {
      // issue async B copy for next tile immediately (overlaps with WMMA below);
      // safe: buffer `nxt` readers all passed the barrier ending the previous iter
      const u16* bs = bSrcBase + (kt + 1) * BK;
      const u32 boff = lds_off(&Bs[nxt][crow * LDB + cseg]);
      async_copy_b128(boff,      bs);
      async_copy_b128(boff + 16, bs + 8);
      // A: fp32 global -> regs
      const float4* s = reinterpret_cast<const float4*>(aSrcBase + (kt + 1) * BK);
      f0 = s[0]; f1 = s[1]; f2 = s[2]; f3 = s[3];
      if (kt + 2 < KT) {
        __builtin_prefetch(aSrcBase + (kt + 2) * BK, 0, 1);
        __builtin_prefetch(bSrcBase + (kt + 2) * BK, 0, 1);
      }
    }

    // fragment loads from LDS (conflict-free ds_load_b128)
    union FragBF { v16bf v; uint4 q[2]; };
    FragBF fa[2], fb[4];
    const u16* Abase = &As[cur][0];
    const u16* Bbase = &Bs[cur][0];
#pragma unroll
    for (int mi = 0; mi < 2; ++mi) {
      const int arow = waveM * 32 + mi * 16 + ln;
      const u16* p = Abase + arow * LDA + half * 8;   // K {0..7 | 8..15}, {16..23 | 24..31}
      fa[mi].q[0] = *reinterpret_cast<const uint4*>(p);
      fa[mi].q[1] = *reinterpret_cast<const uint4*>(p + 16);
    }
#pragma unroll
    for (int ni = 0; ni < 4; ++ni) {
      const int bcol = waveN * 64 + ni * 16 + ln;
      const u16* p = Bbase + bcol * LDB + half * 8;   // K {0..7 | 8..15}, {16..23 | 24..31}
      fb[ni].q[0] = *reinterpret_cast<const uint4*>(p);
      fb[ni].q[1] = *reinterpret_cast<const uint4*>(p + 16);
    }

#pragma unroll
    for (int mi = 0; mi < 2; ++mi)
#pragma unroll
      for (int ni = 0; ni < 4; ++ni)
        acc[mi][ni] = __builtin_amdgcn_wmma_f32_16x16x32_bf16(
            false, fa[mi].v, false, fb[ni].v, (short)0, acc[mi][ni], false, false);

    if (hasNext) {
      uint4 p0, p1;
      p0.x = pack_bf16(f0.x, f0.y); p0.y = pack_bf16(f0.z, f0.w);
      p0.z = pack_bf16(f1.x, f1.y); p0.w = pack_bf16(f1.z, f1.w);
      p1.x = pack_bf16(f2.x, f2.y); p1.y = pack_bf16(f2.z, f2.w);
      p1.z = pack_bf16(f3.x, f3.y); p1.w = pack_bf16(f3.z, f3.w);
      u16* d = &As[nxt][crow * LDA + cseg];
      *reinterpret_cast<uint4*>(d)     = p0;
      *reinterpret_cast<uint4*>(d + 8) = p1;
      wait_asynccnt0();       // our async B stores to `nxt` are done
      __syncthreads();
    }
  }

  // epilogue: C layout — VGPR r holds M = r + 8*half (lanes split), N = ln
#pragma unroll
  for (int mi = 0; mi < 2; ++mi) {
#pragma unroll
    for (int r = 0; r < 8; ++r) {
      const int gm = gm0 + waveM * 32 + mi * 16 + half * 8 + r;
      const int4  idx = rp_idx[gm];
      const float4 wt = rp_w[gm];
      const float* p0 = pos + (size_t)idx.x * DDIM;
      const float* p1 = pos + (size_t)idx.y * DDIM;
      const float* p2 = pos + (size_t)idx.z * DDIM;
      const float* p3 = pos + (size_t)idx.w * DDIM;
#pragma unroll
      for (int ni = 0; ni < 4; ++ni) {
        const int gn = gn0 + waveN * 64 + ni * 16 + ln;
        const float v = acc[mi][ni][r] + bias[gn]
                      + wt.x * p0[gn] + wt.y * p1[gn]
                      + wt.z * p2[gn] + wt.w * p3[gn];
        out[(size_t)gm * DDIM + gn] = v;
      }
    }
  }
}

extern "C" void kernel_launch(void* const* d_in, const int* in_sizes, int n_in,
                              void* d_out, int out_size, void* d_ws, size_t ws_size,
                              hipStream_t stream) {
  (void)in_sizes; (void)n_in; (void)out_size; (void)ws_size;
  const float* pix  = (const float*)d_in[0];  // [64,1024,768] fp32
  const int*   ss   = (const int*)d_in[1];    // [64,2] int32
  const float* Wf   = (const float*)d_in[2];  // [1152,768] fp32
  const float* bias = (const float*)d_in[3];  // [1152] fp32
  const float* pos  = (const float*)d_in[4];  // [256,1152] fp32
  float* out = (float*)d_out;                 // [64,1024,1152] fp32

  char* ws = (char*)d_ws;
  u32*    Wbf   = (u32*)ws;                              // 1,769,472 B of bf16 W
  int4*   rpIdx = (int4*)(ws + 1769472);                 // 1,048,576 B
  float4* rpW   = (float4*)(ws + 1769472 + 1048576);     // 1,048,576 B

  prep_w_kernel  <<<432, 256, 0, stream>>>(Wf, Wbf);
  prep_pos_kernel<<<256, 256, 0, stream>>>(ss, rpIdx, rpW);
  // N-tiles fastest (x), M-tiles slow (y): A bands stay L2-hot across the 9 N-blocks
  embed_gemm_kernel<<<dim3(9, 512), 256, 0, stream>>>(
      pix, (const u16*)Wbf, bias, pos, rpIdx, rpW, out);
}
